// PDiscoModel_11201274708047
// MI455X (gfx1250) — compile-verified, matrix operand order checked by reading, not run on hardware
//
#include <hip/hip_runtime.h>
#include <hip/hip_bf16.h>

typedef __attribute__((ext_vector_type(16))) __bf16 v16bf;
typedef __attribute__((ext_vector_type(8)))  __bf16 v8bf;
typedef __attribute__((ext_vector_type(8)))  float  v8f;

#define B_   32
#define P_   576
#define D_   768
#define K_   16
#define KP1  17
#define C_   200
#define CP_  208   // C padded to 13 tiles of 16

// ---------------------------------------------------------------------------
// Fragment loader: lane l holds 16 bf16 K-values as two contiguous 8-element
// chunks at byte-aligned offsets o0 = 8*(l>>4) and o1 = 16 + 8*(l>>4)
// (matches the 16-bit A/B VGPR layout in CDNA5 ISA §7.12.2).
// ---------------------------------------------------------------------------
__device__ __forceinline__ v16bf load_frag(const __bf16* base, int o0, int o1) {
  v8bf lo = *(const v8bf*)(base + o0);
  v8bf hi = *(const v8bf*)(base + o1);
  v16bf r;
#pragma unroll
  for (int i = 0; i < 8; ++i) { r[i] = lo[i]; r[i + 8] = hi[i]; }
  return r;
}

// ---------------------------------------------------------------------------
// Kernel 0a: x (fp32 row-major [b*p][d]) -> bf16 copy + per-row sum of squares
// ---------------------------------------------------------------------------
__global__ __launch_bounds__(256)
void k_rows(const float* __restrict__ x, __bf16* __restrict__ xb,
            float* __restrict__ x_sq) {
  const int row = blockIdx.x;
  const int t = threadIdx.x;
  float ss = 0.f;
  for (int i = t; i < D_; i += 256) {
    float v = x[(size_t)row * D_ + i];
    xb[(size_t)row * D_ + i] = (__bf16)v;
    ss += v * v;
  }
  __shared__ float red[256];
  red[t] = ss; __syncthreads();
  for (int s = 128; s > 0; s >>= 1) {
    if (t < s) red[t] += red[t + s];
    __syncthreads();
  }
  if (t == 0) x_sq[row] = red[0];
}

// ---------------------------------------------------------------------------
// Kernel 0b: prototypes (17 x 768) -> bf16, padded to 32 rows; p_sq per row
// ---------------------------------------------------------------------------
__global__ __launch_bounds__(256)
void k_proto(const float* __restrict__ proto, __bf16* __restrict__ pb,
             float* __restrict__ p_sq) {
  const int r = blockIdx.x;            // 0..31 (rows 17..31 zero-filled)
  const int t = threadIdx.x;
  float ss = 0.f;
  for (int i = t; i < D_; i += 256) {
    float v = (r < KP1) ? proto[(size_t)r * D_ + i] : 0.f;
    pb[(size_t)r * D_ + i] = (__bf16)v;
    ss += v * v;
  }
  __shared__ float red[256];
  red[t] = ss; __syncthreads();
  for (int s = 128; s > 0; s >>= 1) {
    if (t < s) red[t] += red[t + s];
    __syncthreads();
  }
  if (t == 0 && r < KP1) p_sq[r] = red[0];
}

// ---------------------------------------------------------------------------
// Kernel 0c: x -> xT bf16 ([b][d][p]) via LDS 32x32 tiles, coalesced both ways
// ---------------------------------------------------------------------------
__global__ __launch_bounds__(256)
void k_transpose(const float* __restrict__ x, __bf16* __restrict__ xT) {
  __shared__ float tile[32][33];
  const int b  = blockIdx.z;
  const int p0 = blockIdx.x * 32;
  const int d0 = blockIdx.y * 32;
  const int tx = threadIdx.x & 31;
  const int ty = threadIdx.x >> 5;   // 0..7
  for (int r = ty; r < 32; r += 8)
    tile[r][tx] = x[((size_t)b * P_ + p0 + r) * D_ + d0 + tx];
  __syncthreads();
  for (int r = ty; r < 32; r += 8)
    xT[((size_t)b * D_ + d0 + r) * P_ + p0 + tx] = (__bf16)tile[tx][r];
}

// ---------------------------------------------------------------------------
// Kernel 0d: w_cls (768 x 200, d-major) -> w_clsT bf16 (208 x 768, c-major)
// ---------------------------------------------------------------------------
__global__ __launch_bounds__(256)
void k_wclsT(const float* __restrict__ w, __bf16* __restrict__ wT) {
  const int c = blockIdx.x;            // 0..207
  const int t = threadIdx.x;
  for (int i = t; i < D_; i += 256) {
    float v = (c < C_) ? w[(size_t)i * C_ + c] : 0.f;
    wT[(size_t)c * D_ + i] = (__bf16)v;
  }
}

// ---------------------------------------------------------------------------
// Kernel 1: xpT GEMM (M=k pad 32, N=16 p's, K=768) + fused softmax over k.
// One wave per block. Writes A output (b,17,p) and a_bf in [b][k][p] layout.
// ---------------------------------------------------------------------------
__global__ __launch_bounds__(32)
void k_xp_softmax(const __bf16* __restrict__ xb, const __bf16* __restrict__ pb,
                  const float* __restrict__ x_sq, const float* __restrict__ p_sq,
                  float* __restrict__ outA, __bf16* __restrict__ a_bf) {
  const int b    = blockIdx.y;
  const int p0   = blockIdx.x * 16;
  const int l    = threadIdx.x;
  const int lane = l & 15;
  const int h    = l >> 4;
  const int o0 = 8 * h, o1 = 16 + 8 * h;

  const __bf16* arow0 = pb + (size_t)lane * D_;          // proto rows 0..15
  const __bf16* arow1 = pb + (size_t)(16 + lane) * D_;   // proto rows 16..31
  const __bf16* brow  = xb + ((size_t)b * P_ + p0 + lane) * D_;

  v8f acc0 = {}; v8f acc1 = {};
#pragma unroll 4
  for (int d0 = 0; d0 < D_; d0 += 32) {
    v16bf bf = load_frag(brow + d0, o0, o1);
    v16bf a0 = load_frag(arow0 + d0, o0, o1);
    v16bf a1 = load_frag(arow1 + d0, o0, o1);
    acc0 = __builtin_amdgcn_wmma_f32_16x16x32_bf16(false, a0, false, bf,
                                                   (short)0, acc0, false, false);
    acc1 = __builtin_amdgcn_wmma_f32_16x16x32_bf16(false, a1, false, bf,
                                                   (short)0, acc1, false, false);
  }

  // Score = -dists = 2*xp - x_sq - p_sq, dumped to LDS for per-column softmax.
  __shared__ float sc[KP1][16];
  const float xs = x_sq[(size_t)b * P_ + p0 + lane];
#pragma unroll
  for (int r = 0; r < 8; ++r) {
    int m = r + 8 * h;                           // k index 0..15
    sc[m][lane] = 2.0f * acc0[r] - xs - p_sq[m];
  }
  if (h == 0)                                     // k = 16 lives in acc1[0], lanes 0..15
    sc[16][lane] = 2.0f * acc1[0] - xs - p_sq[16];
  __syncthreads();

  if (l < 16) {
    const int p = p0 + l;
    float mx = -1e30f;
#pragma unroll
    for (int k = 0; k < KP1; ++k) mx = fmaxf(mx, sc[k][l]);
    float e[KP1], sum = 0.f;
#pragma unroll
    for (int k = 0; k < KP1; ++k) { e[k] = __expf(sc[k][l] - mx); sum += e[k]; }
    const float inv = 1.f / sum;
#pragma unroll
    for (int k = 0; k < KP1; ++k) {
      float a = e[k] * inv;
      outA[((size_t)b * KP1 + k) * P_ + p] = a;          // A already transposed
      if (k < K_)
        a_bf[((size_t)b * K_ + k) * P_ + p] = (__bf16)a; // GEMM2 A operand
    }
  }
}

// ---------------------------------------------------------------------------
// Kernel 2: v_fg = a^T x / P. M=16 (k), N=768 (d, 48 tiles), K=576 (p).
// ---------------------------------------------------------------------------
__global__ __launch_bounds__(32)
void k_vfg(const __bf16* __restrict__ a_bf, const __bf16* __restrict__ xT,
           float* __restrict__ v_fg) {
  const int b  = blockIdx.y;
  const int nt = blockIdx.x;                       // d tile 0..47
  const int l = threadIdx.x;
  const int lane = l & 15, h = l >> 4;
  const int o0 = 8 * h, o1 = 16 + 8 * h;

  const __bf16* arow = a_bf + ((size_t)b * K_ + lane) * P_;
  const __bf16* brow = xT   + ((size_t)b * D_ + nt * 16 + lane) * P_;

  v8f acc = {};
#pragma unroll 2
  for (int pp = 0; pp < P_; pp += 32) {
    v16bf af = load_frag(arow + pp, o0, o1);
    v16bf bf = load_frag(brow + pp, o0, o1);
    acc = __builtin_amdgcn_wmma_f32_16x16x32_bf16(false, af, false, bf,
                                                  (short)0, acc, false, false);
  }
  const float inv = 1.0f / (float)P_;
  const int d = nt * 16 + lane;
#pragma unroll
  for (int r = 0; r < 8; ++r) {
    int m = r + 8 * h;                             // k index
    v_fg[((size_t)b * K_ + m) * D_ + d] = acc[r] * inv;  // coalesced 64B stores
  }
}

// ---------------------------------------------------------------------------
// Kernel 3: LayerNorm over D per (b,k); emits fp32 v_norm output + bf16 copy.
// ---------------------------------------------------------------------------
__global__ __launch_bounds__(256)
void k_ln(const float* __restrict__ v_fg, const float* __restrict__ gamma,
          const float* __restrict__ beta, float* __restrict__ out_vnorm,
          __bf16* __restrict__ vb) {
  const int row = blockIdx.x;                      // b*16 + k
  const int t = threadIdx.x;
  const float* v = v_fg + (size_t)row * D_;
  float s = 0.f, s2 = 0.f;
  for (int i = t; i < D_; i += 256) { float x = v[i]; s += x; s2 += x * x; }
  __shared__ float rs[256], rq[256];
  rs[t] = s; rq[t] = s2; __syncthreads();
  for (int k = 128; k > 0; k >>= 1) {
    if (t < k) { rs[t] += rs[t + k]; rq[t] += rq[t + k]; }
    __syncthreads();
  }
  const float mu  = rs[0] / (float)D_;
  const float var = rq[0] / (float)D_ - mu * mu;
  const float inv = rsqrtf(var + 1e-6f);
  for (int i = t; i < D_; i += 256) {
    float y = (v[i] - mu) * inv * gamma[i] + beta[i];
    out_vnorm[(size_t)row * D_ + i] = y;
    vb[(size_t)row * D_ + i] = (__bf16)y;
  }
}

// ---------------------------------------------------------------------------
// Kernel 4: logits_parts = v_norm @ w_cls + b_cls. M=512 (32 tiles), N=208
// (13 tiles, c>=200 masked in epilogue), K=768.
// ---------------------------------------------------------------------------
__global__ __launch_bounds__(32)
void k_logits(const __bf16* __restrict__ vb, const __bf16* __restrict__ wT,
              const float* __restrict__ b_cls, float* __restrict__ out_parts) {
  const int mt = blockIdx.y;                       // 0..31
  const int nt = blockIdx.x;                       // 0..12
  const int l = threadIdx.x;
  const int lane = l & 15, h = l >> 4;
  const int o0 = 8 * h, o1 = 16 + 8 * h;
  const __bf16* arow = vb + ((size_t)mt * 16 + lane) * D_;
  const __bf16* brow = wT + ((size_t)nt * 16 + lane) * D_;

  v8f acc = {};
#pragma unroll 4
  for (int d0 = 0; d0 < D_; d0 += 32) {
    v16bf af = load_frag(arow + d0, o0, o1);
    v16bf bf = load_frag(brow + d0, o0, o1);
    acc = __builtin_amdgcn_wmma_f32_16x16x32_bf16(false, af, false, bf,
                                                  (short)0, acc, false, false);
  }
  const int c = nt * 16 + lane;
  if (c < C_) {
    const float bias = b_cls[c];
#pragma unroll
    for (int r = 0; r < 8; ++r) {
      int row = mt * 16 + r + 8 * h;               // b*16 + k
      out_parts[(size_t)row * C_ + c] = acc[r] + bias;
    }
  }
}

// ---------------------------------------------------------------------------
// Kernel 5: logits_agg = mean over k of logits_parts.
// ---------------------------------------------------------------------------
__global__ __launch_bounds__(256)
void k_agg(const float* __restrict__ parts, float* __restrict__ agg) {
  const int b = blockIdx.x;
  const int c = threadIdx.x;
  if (c < C_) {
    float s = 0.f;
    for (int k = 0; k < K_; ++k) s += parts[((size_t)b * K_ + k) * C_ + c];
    agg[(size_t)b * C_ + c] = s * (1.0f / (float)K_);
  }
}

// ---------------------------------------------------------------------------
extern "C" void kernel_launch(void* const* d_in, const int* in_sizes, int n_in,
                              void* d_out, int out_size, void* d_ws, size_t ws_size,
                              hipStream_t stream) {
  (void)in_sizes; (void)n_in; (void)out_size; (void)ws_size;

  const float* x      = (const float*)d_in[0];  // (32,24,24,768)
  const float* proto  = (const float*)d_in[1];  // (17,768)
  const float* gamma  = (const float*)d_in[2];  // (768)
  const float* beta   = (const float*)d_in[3];  // (768)
  const float* w_cls  = (const float*)d_in[4];  // (768,200)
  const float* b_cls  = (const float*)d_in[5];  // (200)

  float* out  = (float*)d_out;
  float* outA = out;                                   // 32*17*576
  float* outV = outA + (size_t)B_ * KP1 * P_;          // 32*16*768
  float* outP = outV + (size_t)B_ * K_ * D_;           // 32*16*200
  float* outG = outP + (size_t)B_ * K_ * C_;           // 32*200

  // Workspace carve (256B aligned slices)
  char* ws = (char*)d_ws;
  size_t off = 0;
  auto take = [&](size_t bytes) -> char* {
    char* p = ws + off;
    off = (off + bytes + 255) & ~(size_t)255;
    return p;
  };
  __bf16* x_bf   = (__bf16*)take((size_t)B_ * P_ * D_ * 2);   // 28.3 MB
  __bf16* xT_bf  = (__bf16*)take((size_t)B_ * D_ * P_ * 2);   // 28.3 MB
  float*  x_sq   = (float*) take((size_t)B_ * P_ * 4);
  __bf16* pb     = (__bf16*)take((size_t)32 * D_ * 2);
  float*  p_sq   = (float*) take((size_t)32 * 4);
  __bf16* a_bf   = (__bf16*)take((size_t)B_ * K_ * P_ * 2);
  float*  v_fg   = (float*) take((size_t)B_ * K_ * D_ * 4);
  __bf16* vn_bf  = (__bf16*)take((size_t)B_ * K_ * D_ * 2);
  __bf16* wT_bf  = (__bf16*)take((size_t)CP_ * D_ * 2);

  // 0: precision conversion / transposes / row reductions
  k_rows     <<<dim3(B_ * P_),       dim3(256), 0, stream>>>(x, x_bf, x_sq);
  k_proto    <<<dim3(32),            dim3(256), 0, stream>>>(proto, pb, p_sq);
  k_transpose<<<dim3(P_/32, D_/32, B_), dim3(256), 0, stream>>>(x, xT_bf);
  k_wclsT    <<<dim3(CP_),           dim3(256), 0, stream>>>(w_cls, wT_bf);

  // 1: distances via WMMA + fused softmax -> A output + a (bf16, [b][k][p])
  k_xp_softmax<<<dim3(P_/16, B_),    dim3(32),  0, stream>>>(x_bf, pb, x_sq, p_sq,
                                                             outA, a_bf);
  // 2: v_fg = a^T x / P via WMMA
  k_vfg      <<<dim3(D_/16, B_),     dim3(32),  0, stream>>>(a_bf, xT_bf, v_fg);

  // 3: LayerNorm -> v_norm output (fp32) + bf16 copy for GEMM3
  k_ln       <<<dim3(B_ * K_),       dim3(256), 0, stream>>>(v_fg, gamma, beta,
                                                             outV, vn_bf);
  // 4: logits_parts via WMMA
  k_logits   <<<dim3(CP_/16, B_*K_/16), dim3(32), 0, stream>>>(vn_bf, wT_bf,
                                                               b_cls, outP);
  // 5: logits_agg = mean over k
  k_agg      <<<dim3(B_),            dim3(256), 0, stream>>>(outP, outG);
}
